// ActuatorPolicyNet_41394894799580
// MI455X (gfx1250) — matrix operand order repaired and checked by reading, not compiled, not run on hardware
//
#include <hip/hip_runtime.h>
#include <hip/hip_bf16.h>
#include <math.h>

// ---------------------------------------------------------------------------
// Types for CDNA5 WMMA (wave32, 16x16x32 bf16 -> f32)
// ---------------------------------------------------------------------------
typedef __bf16 bf16;
typedef __attribute__((ext_vector_type(16))) __bf16 v16bf;
typedef __attribute__((ext_vector_type(8)))  __bf16 v8bf;
typedef __attribute__((ext_vector_type(8)))  float  v8f;
typedef __attribute__((ext_vector_type(4)))  int    v4i;

#define SEQ  2048
#define INSZ 512
#define H    1024
#define G4H  (4 * H)

// CDNA5 async global->LDS copy builtins (probe-verified on this toolchain)
#if __has_builtin(__builtin_amdgcn_global_load_async_to_lds_b128) && \
    __has_builtin(__builtin_amdgcn_s_wait_asynccnt)
#define ATH_ASYNC 1
#else
#define ATH_ASYNC 0
#endif

// ===========================================================================
// Small prep kernels
// ===========================================================================
__global__ __launch_bounds__(256) void initKernel(const float* __restrict__ b_ih,
                                                  const float* __restrict__ b_hh,
                                                  const float* __restrict__ h0,
                                                  float* __restrict__ bsum,
                                                  float* __restrict__ hbuf,
                                                  unsigned* __restrict__ bar) {
  int i = blockIdx.x * 256 + threadIdx.x;
  if (i < G4H) bsum[i] = b_ih[i] + b_hh[i];
  if (i < H) { hbuf[i] = h0[i]; hbuf[H + i] = 0.f; }
  if (i < 128) bar[i] = 0u;
}

__global__ __launch_bounds__(256) void convertBf16(const float* __restrict__ in,
                                                   bf16* __restrict__ out, int n) {
  int i = blockIdx.x * 256 + threadIdx.x;
  if (i < n) out[i] = (bf16)in[i];
}

// Pack W [N x K] f32 (row-major, out_features x in_features) into the WMMA
// B-fragment global layout for B = W^T [K x N]:
//   Bf[(((k/32)*(N/16) + n/16)*32 + lane)*16 + e] = W[n][k]
//   where lane = (hl = (k%32)/16) * 16 + (n%16),  e = k%16.
// Each 16x16x32-K B fragment then is one contiguous 32B run per lane.
__global__ __launch_bounds__(256) void packBFrag(const float* __restrict__ W,
                                                 bf16* __restrict__ Bf,
                                                 int N, int K) {
  size_t idx = (size_t)blockIdx.x * 256 + threadIdx.x;
  if (idx >= (size_t)N * K) return;
  int e = (int)(idx & 15);
  size_t t1 = idx >> 4;
  int lane = (int)(t1 & 31);
  size_t frag = t1 >> 5;                 // kb * nTiles + nb
  int nTiles = N >> 4;
  int nb = (int)(frag % nTiles);
  int kb = (int)(frag / nTiles);
  int n = nb * 16 + (lane & 15);
  int k = kb * 32 + (lane >> 4) * 16 + e;
  Bf[idx] = (bf16)W[(size_t)n * K + k];
}

// ===========================================================================
// WMMA bf16 GEMM, LDS-free: A row-major, B pre-packed in fragment order.
// Block tile 128x128, 8 waves (4x2), each wave 32x64 = 2x4 fragments.
// Per k-step: 4 b128 A-loads + 8 b128 B-loads + 8 v_wmma.
// RELU / output format are compile-time to keep the epilogue branch-free.
// ===========================================================================
template <bool RELU, bool WRITE_F, bool WRITE_B>
__global__ __launch_bounds__(256) void gemmBf16Wmma(const bf16* __restrict__ A,
                                                    const bf16* __restrict__ Bf,
                                                    const float* __restrict__ bias,
                                                    float* __restrict__ Cf,
                                                    bf16* __restrict__ Cb,
                                                    int N, int K) {
  const int tid  = threadIdx.x;
  const int lane = tid & 31;
  const int wid  = tid >> 5;
  const int wm   = wid >> 1;   // 0..3 (M dir)
  const int wn   = wid & 1;    // 0..1 (N dir)
  const int hl   = lane >> 4;
  const int l16  = lane & 15;
  const int m0   = blockIdx.y * 128;
  const int n0   = blockIdx.x * 128;

  v8f acc[2][4] = {};

  const bf16* ar[2];
  ar[0] = A + (size_t)(m0 + wm * 32 + l16) * K;
  ar[1] = A + (size_t)(m0 + wm * 32 + 16 + l16) * K;

  const bf16* bp[4];
#pragma unroll
  for (int j = 0; j < 4; ++j) {
    int nb = (n0 >> 4) + wn * 4 + j;
    bp[j] = Bf + ((size_t)nb * 32 + lane) * 16;
  }
  const size_t bstep = (size_t)(N >> 4) * 32 * 16;  // elements per kb step

  for (int kk = 0; kk < K; kk += 32) {
    if (lane == 0 && kk + 64 < K) {  // gfx1250 global_prefetch hint
      __builtin_prefetch(&ar[0][kk + 64], 0, 1);
      __builtin_prefetch(bp[0] + 2 * bstep, 0, 1);
    }
    v16bf af[2], bv[4];
#pragma unroll
    for (int i = 0; i < 2; ++i) {
      union { v16bf v; v8bf h[2]; } u;
      u.h[0] = *(const v8bf*)&ar[i][kk + hl * 16];
      u.h[1] = *(const v8bf*)&ar[i][kk + hl * 16 + 8];
      af[i] = u.v;
    }
#pragma unroll
    for (int j = 0; j < 4; ++j) {
      bv[j] = *(const v16bf*)bp[j];
      bp[j] += bstep;
    }
#pragma unroll
    for (int i = 0; i < 2; ++i)
#pragma unroll
      for (int j = 0; j < 4; ++j)
        acc[i][j] = __builtin_amdgcn_wmma_f32_16x16x32_bf16(
            false, af[i], false, bv[j], (short)0, acc[i][j], false, false);
  }

  // Branch-free epilogue: bias (+relu), write f32 and/or bf16
#pragma unroll
  for (int i = 0; i < 2; ++i) {
#pragma unroll
    for (int j = 0; j < 4; ++j) {
      int nglob = n0 + wn * 64 + j * 16 + l16;
      float bvv = bias[nglob];
#pragma unroll
      for (int v = 0; v < 8; ++v) {
        int mglob = m0 + wm * 32 + i * 16 + hl * 8 + v;
        float val = acc[i][j][v] + bvv;
        if (RELU) val = fmaxf(val, 0.f);
        if (WRITE_F) Cf[(size_t)mglob * N + nglob] = val;
        if (WRITE_B) Cb[(size_t)mglob * N + nglob] = (bf16)val;
      }
    }
  }
}

// ===========================================================================
// Persistent LSTM scan. 256 WGs x 256 threads; each WG owns 4 hidden units
// (16 gate rows). W_hh slice (bf16, 32 KB) resident in LDS for all 2048 steps;
// staged via CDNA5 async global->LDS copy (ASYNCcnt) when available.
// ===========================================================================
#define SCAN_WGS 256
#define UPW 4
#define ROWS 16

__device__ __forceinline__ void gridBarrier(unsigned* cnt, unsigned* gen, unsigned nwg) {
  __threadfence();
  __syncthreads();
  if (threadIdx.x == 0) {
    unsigned g = __hip_atomic_load(gen, __ATOMIC_ACQUIRE, __HIP_MEMORY_SCOPE_AGENT);
    if (__hip_atomic_fetch_add(cnt, 1u, __ATOMIC_ACQ_REL, __HIP_MEMORY_SCOPE_AGENT) ==
        nwg - 1u) {
      __hip_atomic_store(cnt, 0u, __ATOMIC_RELAXED, __HIP_MEMORY_SCOPE_AGENT);
      __hip_atomic_fetch_add(gen, 1u, __ATOMIC_RELEASE, __HIP_MEMORY_SCOPE_AGENT);
    } else {
      while (__hip_atomic_load(gen, __ATOMIC_ACQUIRE, __HIP_MEMORY_SCOPE_AGENT) == g)
        __builtin_amdgcn_s_sleep(2);
    }
  }
  __syncthreads();
}

__global__ __launch_bounds__(256) void lstmScan(const bf16* __restrict__ whhb, // [4H][H] bf16
                                                const float* __restrict__ gx,
                                                const float* __restrict__ c0,
                                                float* __restrict__ hbuf,   // [2][H]
                                                float* __restrict__ cstate, // [H]
                                                bf16* __restrict__ ysb,     // [SEQ][H]
                                                unsigned* cnt, unsigned* gen) {
  __shared__ __align__(16) bf16 wlds[ROWS][H];   // 32 KB resident W_hh slice
  __shared__ float gsh[ROWS];

  const int t0 = threadIdx.x;
  const int u0 = blockIdx.x * UPW;

  // Stage this WG's 16 gate rows (bf16) into LDS: 2048 chunks of 16 bytes.
  for (int c = t0; c < ROWS * H / 8; c += 256) {
    int r = c >> 7, kc = c & 127;
    int gr = (r >> 2) * H + u0 + (r & 3);
#if ATH_ASYNC
    __builtin_amdgcn_global_load_async_to_lds_b128(
        (v4i*)&whhb[(size_t)gr * H + kc * 8],
        (__attribute__((address_space(3))) v4i*)&wlds[r][kc * 8],
        0, 0);
#else
    *(v8bf*)&wlds[r][kc * 8] = *(const v8bf*)&whhb[(size_t)gr * H + kc * 8];
#endif
  }
#if ATH_ASYNC
  __builtin_amdgcn_s_wait_asynccnt(0);
#endif
  float creg = 0.f;
  if (t0 < UPW) creg = c0[u0 + t0];
  __syncthreads();

  const int row   = t0 >> 4;   // 0..15 local gate row
  const int chunk = t0 & 15;   // 64-wide k slice

  for (int t = 0; t < SEQ; ++t) {
    const float* h = hbuf + (t & 1) * H;
    float acc = 0.f;
    const bf16*  wr = &wlds[row][chunk * 64];
    const float* hc = h + chunk * 64;
#pragma unroll 8
    for (int k = 0; k < 64; ++k) acc += (float)wr[k] * hc[k];
    for (int off = 8; off; off >>= 1) acc += __shfl_xor(acc, off, 32);
    if (chunk == 0) gsh[row] = acc;
    __syncthreads();

    if (t0 < UPW) {
      int uu = t0;
      const float* g = gx + (size_t)t * G4H + u0 + uu;
      float gi = g[0 * H] + gsh[0 * UPW + uu];
      float gf = g[1 * H] + gsh[1 * UPW + uu];
      float gg = g[2 * H] + gsh[2 * UPW + uu];
      float go = g[3 * H] + gsh[3 * UPW + uu];
      float i_ = 1.f / (1.f + __expf(-gi));
      float f_ = 1.f / (1.f + __expf(-gf));
      float g_ = tanhf(gg);
      float o_ = 1.f / (1.f + __expf(-go));
      creg = f_ * creg + i_ * g_;
      float hn = o_ * tanhf(creg);
      hbuf[((t + 1) & 1) * H + u0 + uu] = hn;
      ysb[(size_t)t * H + u0 + uu] = (bf16)hn;
      if (t == SEQ - 1) cstate[u0 + uu] = creg;
    }
    gridBarrier(cnt, gen, SCAN_WGS);
  }
}

// ===========================================================================
// Head: mean = h2 @ mean_w^T + mean_b ; broadcast log_std
// ===========================================================================
__global__ __launch_bounds__(128) void headKernel(const float* __restrict__ h2,
                                                  const float* __restrict__ mw,
                                                  const float* __restrict__ mb,
                                                  const float* __restrict__ lstd,
                                                  float* __restrict__ out) {
  __shared__ float red[4][128];
  int t = blockIdx.x, j = threadIdx.x;
  float p0 = 0, p1 = 0, p2 = 0, p3 = 0;
  for (int k = j; k < H; k += 128) {
    float hv = h2[(size_t)t * H + k];
    p0 += hv * mw[k];
    p1 += hv * mw[H + k];
    p2 += hv * mw[2 * H + k];
    p3 += hv * mw[3 * H + k];
  }
  red[0][j] = p0; red[1][j] = p1; red[2][j] = p2; red[3][j] = p3;
  __syncthreads();
  for (int s = 64; s > 0; s >>= 1) {
    if (j < s) for (int o = 0; o < 4; ++o) red[o][j] += red[o][j + s];
    __syncthreads();
  }
  if (j < 4) {
    out[t * 4 + j] = red[j][0] + mb[j];
    out[SEQ * 4 + t * 4 + j] = lstd[j];
  }
}

__global__ __launch_bounds__(256) void packKernel(const float* __restrict__ hbuf,
                                                  const float* __restrict__ cstate,
                                                  float* __restrict__ out) {
  int i = blockIdx.x * 256 + threadIdx.x;
  if (i < H) {
    out[2 * SEQ * 4 + i] = hbuf[i];          // hT (final h lives in buffer 0)
    out[2 * SEQ * 4 + H + i] = cstate[i];    // cT
  }
}

// ===========================================================================
// Launch
// ===========================================================================
extern "C" void kernel_launch(void* const* d_in, const int* in_sizes, int n_in,
                              void* d_out, int out_size, void* d_ws, size_t ws_size,
                              hipStream_t stream) {
  (void)in_sizes; (void)n_in; (void)out_size; (void)ws_size;
  const float* x     = (const float*)d_in[0];
  const float* h0    = (const float*)d_in[1];
  const float* c0    = (const float*)d_in[2];
  const float* w_ih  = (const float*)d_in[3];
  const float* w_hh  = (const float*)d_in[4];
  const float* b_ih  = (const float*)d_in[5];
  const float* b_hh  = (const float*)d_in[6];
  const float* fc1_w = (const float*)d_in[7];
  const float* fc1_b = (const float*)d_in[8];
  const float* fc2_w = (const float*)d_in[9];
  const float* fc2_b = (const float*)d_in[10];
  const float* mean_w = (const float*)d_in[11];
  const float* mean_b = (const float*)d_in[12];
  const float* log_std = (const float*)d_in[13];
  float* out = (float*)d_out;

  char* wsp = (char*)d_ws;
  size_t off = 0;
  auto alloc = [&](size_t bytes) -> void* {
    void* p = wsp + off;
    off += (bytes + 255) & ~(size_t)255;
    return p;
  };
  bf16*  xb   = (bf16*)alloc((size_t)SEQ * INSZ * 2);
  bf16*  wihF = (bf16*)alloc((size_t)INSZ * G4H * 2);
  bf16*  fc1F = (bf16*)alloc((size_t)H * H * 2);
  bf16*  fc2F = (bf16*)alloc((size_t)H * H * 2);
  bf16*  whhb = (bf16*)alloc((size_t)G4H * H * 2);
  float* bsum = (float*)alloc((size_t)G4H * 4);
  float* hbuf = (float*)alloc((size_t)2 * H * 4);
  float* cst  = (float*)alloc((size_t)H * 4);
  unsigned* bar = (unsigned*)alloc(512);
  float* gx   = (float*)alloc((size_t)SEQ * G4H * 4);
  bf16*  ysb  = (bf16*)alloc((size_t)SEQ * H * 2);
  bf16*  h1b  = (bf16*)alloc((size_t)SEQ * H * 2);
  float* h2f  = (float*)alloc((size_t)SEQ * H * 4);

  initKernel<<<16, 256, 0, stream>>>(b_ih, b_hh, h0, bsum, hbuf, bar);
  convertBf16<<<(SEQ * INSZ + 255) / 256, 256, 0, stream>>>(x, xb, SEQ * INSZ);
  convertBf16<<<(G4H * H + 255) / 256, 256, 0, stream>>>(w_hh, whhb, G4H * H);
  packBFrag<<<(G4H * INSZ + 255) / 256, 256, 0, stream>>>(w_ih, wihF, G4H, INSZ);
  packBFrag<<<(H * H + 255) / 256, 256, 0, stream>>>(fc1_w, fc1F, H, H);
  packBFrag<<<(H * H + 255) / 256, 256, 0, stream>>>(fc2_w, fc2F, H, H);

  // gx = x @ W_ih^T + (b_ih + b_hh)   (f32 out, no relu)
  gemmBf16Wmma<false, true, false><<<dim3(G4H / 128, SEQ / 128), 256, 0, stream>>>(
      xb, wihF, bsum, gx, nullptr, G4H, INSZ);

  // sequential scan with LDS-resident W_hh
  lstmScan<<<SCAN_WGS, 256, 0, stream>>>(whhb, gx, c0, hbuf, cst, ysb, bar, bar + 64);

  // h1 = relu(ys @ fc1^T + b1) -> bf16 ; h2 = relu(h1 @ fc2^T + b2) -> f32
  gemmBf16Wmma<true, false, true><<<dim3(H / 128, SEQ / 128), 256, 0, stream>>>(
      ysb, fc1F, fc1_b, nullptr, h1b, H, H);
  gemmBf16Wmma<true, true, false><<<dim3(H / 128, SEQ / 128), 256, 0, stream>>>(
      h1b, fc2F, fc2_b, h2f, nullptr, H, H);

  headKernel<<<SEQ, 128, 0, stream>>>(h2f, mean_w, mean_b, log_std, out);
  packKernel<<<4, 256, 0, stream>>>(hbuf, cst, out);
}
